// FAVOR_45526653338119
// MI455X (gfx1250) — matrix operand order repaired and checked by reading, not compiled, not run on hardware
//
#include <hip/hip_runtime.h>
#include <hip/hip_bf16.h>

typedef __bf16 v16bf __attribute__((ext_vector_type(16)));
typedef __bf16 v8bf  __attribute__((ext_vector_type(8)));
typedef __bf16 v4bf  __attribute__((ext_vector_type(4)));
typedef float  v8f   __attribute__((ext_vector_type(8)));

#define NB 16
#define ND 128
#define NL 8192
#define NM 128
#define NC 16            // L chunks for phase 1
#define CHUNK (NL / NC)  // 512
#define LSTEP 32

// padded LDS strides (elements). 136*2B=272B (68 dw, gcd 4 with 64 banks),
// 40*2B=80B (20 dw), 132*4B=528B (132 dw) -> conflict-free lane strides,
// all multiples of 16B for ds_load_b128 rows.
#define SW 136   // lds_w   [m][d]
#define SK 136   // lds_k/q [l'][d]
#define SV 40    // lds_v   [vd][l'], lds_phi(k1) [m][l']
#define SP 136   // lds_phi(k3) [l'][m], lds_kv [vd][m]
#define SO 132   // lds_out [vd][l'] (float)

#define PHI_SCALE 0.08838834764831845f       // 1/sqrt(128)
#define HALF_SCALE 0.044194173824159216f     // 0.5/sqrt(128)

static __device__ inline v8f wmma_bf16(v16bf a, v16bf b, v8f c) {
  return __builtin_amdgcn_wmma_f32_16x16x32_bf16(false, a, false, b, (short)0, c,
                                                 false, false);
}

// relu(x)*PHI_SCALE as (x+|x|)*(PHI_SCALE/2): v_add_f32 (|x| = src modifier) + v_mul
static __device__ inline float relu_scale(float x) {
  return (x + __builtin_fabsf(x)) * HALF_SCALE;
}

// A fragment from row-major [row][k] storage (stride elements, rows 16B-aligned).
// Lane needs k in [kb,kb+16) of row (lane&15), kb = 16*(lane/16): 2x ds_load_b128.
static __device__ inline v16bf a_frag_rm(const __bf16* p, int base, int stride) {
  int lane = threadIdx.x & 31;
  int r = lane & 15;
  int kb = (lane & 16) ? 16 : 0;
  const __bf16* row = p + base + r * stride + kb;
  v8bf lo = *(const v8bf*)(row);
  v8bf hi = *(const v8bf*)(row + 8);
  return __builtin_shufflevector(lo, hi, 0, 1, 2, 3, 4, 5, 6, 7, 8, 9, 10, 11,
                                 12, 13, 14, 15);
}

// B fragment from column-major [n][k] storage: lane needs k in [koff,koff+16)
// of column (lane&15): 2x ds_load_b128.
static __device__ inline v16bf b_frag_cm(const __bf16* p, int base, int stride) {
  int lane = threadIdx.x & 31;
  int n = lane & 15;
  int koff = (lane & 16) ? 16 : 0;
  const __bf16* col = p + base + n * stride + koff;
  v8bf lo = *(const v8bf*)(col);
  v8bf hi = *(const v8bf*)(col + 8);
  return __builtin_shufflevector(lo, hi, 0, 1, 2, 3, 4, 5, 6, 7, 8, 9, 10, 11,
                                 12, 13, 14, 15);
}

static __device__ inline v8f zero8() {
  v8f z = {0.f, 0.f, 0.f, 0.f, 0.f, 0.f, 0.f, 0.f};
  return z;
}

// ---------------------------------------------------------------------------
// Kernel 1: per (batch, L-chunk) -> partial kv (MxDv, fp32) and ksum (M)
// ---------------------------------------------------------------------------
__global__ __launch_bounds__(256) void favor_kv_kernel(
    const float* __restrict__ keys, const float* __restrict__ values,
    const float* __restrict__ features, float* __restrict__ kvpart,
    float* __restrict__ kspart) {
  __shared__ __bf16 lds_w[NM * SW];      // [m][d]   column-major B source
  __shared__ __bf16 lds_k[LSTEP * SK];   // [l'][d]  row-major A source
  __shared__ __bf16 lds_v[ND * SV];      // [vd][l'] column-major B source
  __shared__ __bf16 lds_phi[NM * SV];    // [m][l']  row-major A source

  const int tid = threadIdx.x;
  const int lane = tid & 31;
  const int w = tid >> 5;  // wave 0..7
  const int b = blockIdx.x >> 4;
  const int c = blockIdx.x & 15;

  // stage W transposed: [d][m] fp32 -> [m][d] bf16
  for (int i = tid; i < ND * NM / 4; i += 256) {
    int d = i >> 5;
    int m = (i & 31) << 2;
    float4 f = *(const float4*)&features[(size_t)d * NM + m];
    lds_w[(m + 0) * SW + d] = (__bf16)f.x;
    lds_w[(m + 1) * SW + d] = (__bf16)f.y;
    lds_w[(m + 2) * SW + d] = (__bf16)f.z;
    lds_w[(m + 3) * SW + d] = (__bf16)f.w;
  }
  __syncthreads();

  // Loop-invariant B fragments of W for this wave's m-tile
  v16bf bw[4];
#pragma unroll
  for (int ks = 0; ks < 4; ++ks)
    bw[ks] = b_frag_cm(lds_w, 16 * w * SW + 32 * ks, SW);

  v16bf ones;
#pragma unroll
  for (int j = 0; j < 16; ++j) ones[j] = (__bf16)1.0f;

  v8f kvacc[8];
#pragma unroll
  for (int t = 0; t < 8; ++t) kvacc[t] = zero8();
  v8f ksacc = zero8();

  for (int step = 0; step < CHUNK / LSTEP; ++step) {
    const int l0 = c * CHUNK + step * LSTEP;
    // stage K ([l'][d], transposed scatter) and V ([vd][l'], natural, packed)
    for (int i = tid; i < ND * LSTEP / 4; i += 256) {
      int d = i >> 3;
      int lp = (i & 7) << 2;
      size_t g = ((size_t)b * ND + d) * NL + l0 + lp;
      float4 fk = *(const float4*)&keys[g];
      float4 fv = *(const float4*)&values[g];
      lds_k[(lp + 0) * SK + d] = (__bf16)fk.x;
      lds_k[(lp + 1) * SK + d] = (__bf16)fk.y;
      lds_k[(lp + 2) * SK + d] = (__bf16)fk.z;
      lds_k[(lp + 3) * SK + d] = (__bf16)fk.w;
      v4bf pv = {(__bf16)fv.x, (__bf16)fv.y, (__bf16)fv.z, (__bf16)fv.w};
      *(v4bf*)&lds_v[d * SV + lp] = pv;
    }
    // prefetch next step's tiles (global_prefetch_b8)
    if (step + 1 < CHUNK / LSTEP && tid < ND) {
      size_t gn = ((size_t)b * ND + tid) * NL + l0 + LSTEP;
      __builtin_prefetch(&keys[gn], 0, 0);
      __builtin_prefetch(&values[gn], 0, 0);
    }
    __syncthreads();

    // phi_k = relu(K^T W)*scale ; wave w owns m-tile w, l-tiles 0..1
#pragma unroll
    for (int lt = 0; lt < 2; ++lt) {
      // pre-gather all 4 A fragments so ds latency overlaps the WMMA chain
      v16bf a[4];
#pragma unroll
      for (int ks = 0; ks < 4; ++ks)
        a[ks] = a_frag_rm(lds_k, 16 * lt * SK + 32 * ks, SK);
      v8f acc = zero8();
#pragma unroll
      for (int ks = 0; ks < 4; ++ks) acc = wmma_bf16(a[ks], bw[ks], acc);
#pragma unroll
      for (int r = 0; r < 8; ++r) {
        int lp = 16 * lt + r + ((lane & 16) ? 8 : 0);
        int m = 16 * w + (lane & 15);
        lds_phi[m * SV + lp] = (__bf16)relu_scale(acc[r]);
      }
    }
    __syncthreads();

    // kv[m,vd] += phi^T v  (K-dim = l' = 32); pipeline B fragments 1 ahead
    v16bf a2 = a_frag_rm(lds_phi, 16 * w * SV, SV);
    v16bf bv = b_frag_cm(lds_v, 0, SV);
#pragma unroll
    for (int t = 0; t < 8; ++t) {
      v16bf bnext =
          (t < 7) ? b_frag_cm(lds_v, 16 * (t + 1) * SV, SV) : ones;
      kvacc[t] = wmma_bf16(a2, bv, kvacc[t]);
      bv = bnext;
    }
    ksacc = wmma_bf16(a2, bv, ksacc);  // bv == ones here
    __syncthreads();
  }

  // write partials: kvpart[c][b][m][vd], kspart[c][b][m]
#pragma unroll
  for (int t = 0; t < 8; ++t) {
#pragma unroll
    for (int r = 0; r < 8; ++r) {
      int m = 16 * w + r + ((lane & 16) ? 8 : 0);
      int vd = 16 * t + (lane & 15);
      kvpart[(((size_t)c * NB + b) * NM + m) * NM + vd] = kvacc[t][r];
    }
  }
  if ((lane & 15) == 0) {
#pragma unroll
    for (int r = 0; r < 8; ++r) {
      int m = 16 * w + r + ((lane & 16) ? 8 : 0);
      kspart[((size_t)c * NB + b) * NM + m] = ksacc[r];
    }
  }
}

// ---------------------------------------------------------------------------
// Kernel 2: reduce chunk partials; emit kv as bf16 TRANSPOSED [b][vd][m]
// ---------------------------------------------------------------------------
__global__ __launch_bounds__(256) void favor_reduce_kernel(
    const float* __restrict__ kvpart, const float* __restrict__ kspart,
    __bf16* __restrict__ kvb, __bf16* __restrict__ ksb) {
  int e = blockIdx.x * 256 + threadIdx.x;
  if (e < NB * NM * NM) {
    float s = 0.f;
#pragma unroll
    for (int c = 0; c < NC; ++c) s += kvpart[(size_t)c * NB * NM * NM + e];
    int bb = e >> 14;           // [b][m][vd] source index
    int m = (e >> 7) & 127;
    int vd = e & 127;
    kvb[((size_t)bb * NM + vd) * NM + m] = (__bf16)s;  // [b][vd][m]
  }
  if (e < NB * NM) {
    float s = 0.f;
#pragma unroll
    for (int c = 0; c < NC; ++c) s += kspart[(size_t)c * NB * NM + e];
    ksb[e] = (__bf16)s;
  }
}

// ---------------------------------------------------------------------------
// Kernel 3: per (batch, 128-row L tile): phi_q, out = phi_q·kv / (phi_q·ksum)
// ---------------------------------------------------------------------------
__global__ __launch_bounds__(256) void favor_out_kernel(
    const float* __restrict__ queries, const float* __restrict__ features,
    const __bf16* __restrict__ kvb, const __bf16* __restrict__ ksb,
    float* __restrict__ out) {
  __shared__ __bf16 lds_w[NM * SW];     // [m][d]
  __shared__ __bf16 lds_q[128 * SK];    // [l'][d]
  __shared__ __bf16 lds_phi[128 * SP];  // [l'][m]
  __shared__ __bf16 lds_kv[NM * SP];    // [vd][m]
  __shared__ __bf16 lds_ks[NM];
  __shared__ float lds_den[128];
  __shared__ float lds_out[NM * SO];    // [vd][l']

  const int tid = threadIdx.x;
  const int lane = tid & 31;
  const int w = tid >> 5;
  const int b = blockIdx.x >> 6;
  const int l0 = (blockIdx.x & 63) * 128;

  // stage W [m][d], Q tile [l'][d] (both transposed scatter), kv [vd][m] packed
  for (int i = tid; i < ND * NM / 4; i += 256) {
    int d = i >> 5;
    int x4 = (i & 31) << 2;
    float4 fw = *(const float4*)&features[(size_t)d * NM + x4];
    lds_w[(x4 + 0) * SW + d] = (__bf16)fw.x;
    lds_w[(x4 + 1) * SW + d] = (__bf16)fw.y;
    lds_w[(x4 + 2) * SW + d] = (__bf16)fw.z;
    lds_w[(x4 + 3) * SW + d] = (__bf16)fw.w;
    float4 fq = *(const float4*)&queries[((size_t)b * ND + d) * NL + l0 + x4];
    lds_q[(x4 + 0) * SK + d] = (__bf16)fq.x;
    lds_q[(x4 + 1) * SK + d] = (__bf16)fq.y;
    lds_q[(x4 + 2) * SK + d] = (__bf16)fq.z;
    lds_q[(x4 + 3) * SK + d] = (__bf16)fq.w;
  }
  for (int i = tid; i < NM * NM / 8; i += 256) {
    int vd = i >> 4;
    int m8 = (i & 15) << 3;
    v8bf row = *(const v8bf*)&kvb[((size_t)b * NM + vd) * NM + m8];
    *(v8bf*)&lds_kv[vd * SP + m8] = row;
  }
  if (tid < NM) lds_ks[tid] = ksb[b * NM + tid];
  __syncthreads();

  // phi_q: wave w owns l rows 16w..16w+15, all 8 m-tiles
  {
    v8f pacc[8];
#pragma unroll
    for (int mt = 0; mt < 8; ++mt) pacc[mt] = zero8();
#pragma unroll
    for (int ks = 0; ks < 4; ++ks) {
      v16bf a = a_frag_rm(lds_q, 16 * w * SK + 32 * ks, SK);
      v16bf bm = b_frag_cm(lds_w, 32 * ks, SW);
#pragma unroll
      for (int mt = 0; mt < 8; ++mt) {
        v16bf bnext = (mt < 7)
                          ? b_frag_cm(lds_w, 16 * (mt + 1) * SW + 32 * ks, SW)
                          : bm;
        pacc[mt] = wmma_bf16(a, bm, pacc[mt]);
        bm = bnext;
      }
    }
#pragma unroll
    for (int mt = 0; mt < 8; ++mt) {
#pragma unroll
      for (int r = 0; r < 8; ++r) {
        int lp = 16 * w + r + ((lane & 16) ? 8 : 0);
        int m = 16 * mt + (lane & 15);
        lds_phi[lp * SP + m] = (__bf16)relu_scale(pacc[mt][r]);
      }
    }
  }
  __syncthreads();

  // out = phi_q @ kv ; denom = phi_q @ ksum (ksum in column 0 of extra B)
  v8f oacc[8];
#pragma unroll
  for (int t = 0; t < 8; ++t) oacc[t] = zero8();
  v8f dacc = zero8();
  v16bf bzero;
#pragma unroll
  for (int j = 0; j < 16; ++j) bzero[j] = (__bf16)0.f;

#pragma unroll
  for (int ks = 0; ks < 4; ++ks) {
    v16bf a = a_frag_rm(lds_phi, 16 * w * SP + 32 * ks, SP);
    v16bf bk = b_frag_cm(lds_kv, 32 * ks, SP);
#pragma unroll
    for (int t = 0; t < 8; ++t) {
      v16bf bnext = (t < 7)
                        ? b_frag_cm(lds_kv, 16 * (t + 1) * SP + 32 * ks, SP)
                        : bk;
      oacc[t] = wmma_bf16(a, bk, oacc[t]);
      bk = bnext;
    }
    int koff = (lane & 16) ? 16 : 0;
    v8bf klo = *(const v8bf*)&lds_ks[32 * ks + koff];
    v8bf khi = *(const v8bf*)&lds_ks[32 * ks + koff + 8];
    v16bf bd = __builtin_shufflevector(klo, khi, 0, 1, 2, 3, 4, 5, 6, 7, 8, 9,
                                       10, 11, 12, 13, 14, 15);
    bd = ((lane & 15) == 0) ? bd : bzero;
    dacc = wmma_bf16(a, bd, dacc);
  }

  if ((lane & 15) == 0) {
#pragma unroll
    for (int r = 0; r < 8; ++r)
      lds_den[16 * w + r + ((lane & 16) ? 8 : 0)] = dacc[r];
  }
  __syncthreads();

  float rden[8];
#pragma unroll
  for (int r = 0; r < 8; ++r)
    rden[r] = __builtin_amdgcn_rcpf(lds_den[16 * w + r + ((lane & 16) ? 8 : 0)]);
#pragma unroll
  for (int t = 0; t < 8; ++t) {
#pragma unroll
    for (int r = 0; r < 8; ++r) {
      int vd = 16 * t + (lane & 15);
      int lp = 16 * w + r + ((lane & 16) ? 8 : 0);
      lds_out[vd * SO + lp] = oacc[t][r] * rden[r];
    }
  }
  __syncthreads();

  // coalesced store: out is (B, Dv, L)
  for (int i = tid; i < NM * 128 / 4; i += 256) {
    int vd = i >> 5;
    int lp = (i & 31) << 2;
    float4 v = *(const float4*)&lds_out[vd * SO + lp];
    *(float4*)&out[((size_t)b * NM + vd) * NL + l0 + lp] = v;
  }
}

// ---------------------------------------------------------------------------
extern "C" void kernel_launch(void* const* d_in, const int* in_sizes, int n_in,
                              void* d_out, int out_size, void* d_ws, size_t ws_size,
                              hipStream_t stream) {
  const float* keys = (const float*)d_in[0];
  const float* values = (const float*)d_in[1];
  const float* queries = (const float*)d_in[2];
  const float* features = (const float*)d_in[3];
  float* out = (float*)d_out;

  char* ws = (char*)d_ws;
  size_t off = 0;
  float* kvpart = (float*)(ws + off);
  off += (size_t)NC * NB * NM * NM * sizeof(float);  // 16.78 MB
  float* kspart = (float*)(ws + off);
  off += (size_t)NC * NB * NM * sizeof(float);       // 131 KB
  __bf16* kvb = (__bf16*)(ws + off);
  off += (size_t)NB * NM * NM * sizeof(__bf16);      // 512 KB
  __bf16* ksb = (__bf16*)(ws + off);
  off += (size_t)NB * NM * sizeof(__bf16);

  favor_kv_kernel<<<NC * NB, 256, 0, stream>>>(keys, values, features, kvpart,
                                               kspart);
  favor_reduce_kernel<<<(NB * NM * NM + 255) / 256, 256, 0, stream>>>(
      kvpart, kspart, kvb, ksb);
  favor_out_kernel<<<NB * (NL / 128), 256, 0, stream>>>(queries, features, kvb,
                                                        ksb, out);
}
